// TensHash_21792664060366
// MI455X (gfx1250) — 8-bit matrix operand order repaired and checked by reading, compiled, not run on hardware
//
#include <hip/hip_runtime.h>

#define HIDDEN 256
#define IN_SIZE 32
#define ROUNDS 64
#define BATCH 8192

typedef int          v16i  __attribute__((ext_vector_type(16)));
typedef float        v8f   __attribute__((ext_vector_type(8)));
typedef int          i32x4 __attribute__((ext_vector_type(4)));
typedef int          i32x8 __attribute__((ext_vector_type(8)));
typedef unsigned int u32x4 __attribute__((ext_vector_type(4)));

#define FP8_ONE 0x38u

#define LDS_ROW     272
#define LDS_B_BYTES (HIDDEN * LDS_ROW)
#define LDS_A_BYTES (32 * LDS_ROW)


__global__ void __launch_bounds__(256)
mats_to_fp8(const float* __restrict__ mf, unsigned char* __restrict__ m8, int n) {
    int i = blockIdx.x * blockDim.x + threadIdx.x;
    if (i < n) {
        int v = (int)mf[i];
        m8[i] = v ? (unsigned char)(FP8_ONE | ((v >> 31) & 0x80)) : (unsigned char)0;
    }
}

__global__ void __launch_bounds__(256)
unpack_bits_fp8(const int* __restrict__ nonces, unsigned char* __restrict__ x0) {
    int i = blockIdx.x * blockDim.x + threadIdx.x;
    int b = i >> 8;
    int h = i & 255;
    int bit = (nonces[b * IN_SIZE + (h >> 3)] >> (h & 7)) & 1;
    x0[i] = bit ? (unsigned char)FP8_ONE : (unsigned char)0;
}


static __device__ __forceinline__ void tdm_load_2d(
    unsigned lds_off, const void* gaddr,
    unsigned tensor_d0, unsigned tensor_d1, unsigned stride0,
    unsigned tile_d0, unsigned tile_d1)
{
    unsigned long long ga = (unsigned long long)gaddr;

    u32x4 g0;
    g0[0] = 1u;
    g0[1] = lds_off;
    g0[2] = (unsigned)ga;
    g0[3] = (unsigned)(ga >> 32) | (2u << 30);

    i32x8 g1;
    g1[0] = (1 << 20) | (5 << 22) | (3 << 25);
    g1[1] = (int)((tensor_d0 & 0xFFFFu) << 16);
    g1[2] = (int)((tensor_d0 >> 16) | ((tensor_d1 & 0xFFFFu) << 16));
    g1[3] = (int)((tensor_d1 >> 16) | (tile_d0 << 16));
    g1[4] = (int)(tile_d1 & 0xFFFFu);
    g1[5] = (int)stride0;
    g1[6] = 0;
    g1[7] = 0;

    i32x4 z4 = {0, 0, 0, 0};
#if defined(__clang_major__) && __clang_major__ >= 23
    i32x8 z8 = {0, 0, 0, 0, 0, 0, 0, 0};
    __builtin_amdgcn_tensor_load_to_lds(g0, g1, z4, z4, z8, 0);
#else
    __builtin_amdgcn_tensor_load_to_lds(g0, g1, z4, z4, 0);
#endif
}

template <bool FIRST, bool LAST>
__global__ void __launch_bounds__(256)
round_kernel(const unsigned char* __restrict__ Xin,
             const unsigned char* __restrict__ M8,
             const float* __restrict__ noise,
             unsigned char* __restrict__ Xout,
             float* __restrict__ Fout)
{
    __shared__ __align__(16) char smem[LDS_B_BYTES + LDS_A_BYTES];

    const int lane  = threadIdx.x & 31;
    const int wave  = threadIdx.x >> 5;
    const int halfK = lane >> 4;
    const int lid   = lane & 15;

    const int m0  = blockIdx.x * 32 + (wave >> 2) * 16;
    const int n0q = (wave & 3) * 64;

    if (wave == 0) {
        unsigned lds_base = (unsigned)(unsigned long long)(void*)smem;
        tdm_load_2d(lds_base, M8, HIDDEN, HIDDEN, HIDDEN, HIDDEN, HIDDEN);
        tdm_load_2d(lds_base + LDS_B_BYTES,
                    Xin + (size_t)blockIdx.x * 32 * HIDDEN,
                    HIDDEN, BATCH, HIDDEN, HIDDEN, 32);
        __builtin_amdgcn_s_wait_tensorcnt(0);
    }
    __syncthreads();

    const int mloc = (wave >> 2) * 16 + lid;
    const char* arow = smem + LDS_B_BYTES + mloc * LDS_ROW + halfK * 16;
    v16i a[2];
#pragma unroll
    for (int h = 0; h < 2; ++h)
#pragma unroll
        for (int q = 0; q < 2; ++q)
#pragma unroll
            for (int p = 0; p < 4; ++p) {
                int2 t = *(const int2*)(arow + h * 128 + q * 64 + (p >> 1) * 32 + (p & 1) * 8);
                a[h][q * 8 + 2 * p]     = t.x;
                a[h][q * 8 + 2 * p + 1] = t.y;
            }

#pragma unroll
    for (int t = 0; t < 4; ++t) {
        const int n0   = n0q + t * 16;
        const int ncol = n0 + lid;
        const char* brow = smem + ncol * LDS_ROW + halfK * 16;

        v16i bf[2];
#pragma unroll
        for (int h = 0; h < 2; ++h)
#pragma unroll
            for (int j = 0; j < 4; ++j) {
                int4 u = *(const int4*)(brow + h * 128 + j * 32);
                bf[h][4 * j]     = u.x;
                bf[h][4 * j + 1] = u.y;
                bf[h][4 * j + 2] = u.z;
                bf[h][4 * j + 3] = u.w;
            }

        v8f c = {0.f, 0.f, 0.f, 0.f, 0.f, 0.f, 0.f, 0.f};
        c = __builtin_amdgcn_wmma_f32_16x16x128_fp8_fp8(a[0], bf[0], (short)0, c,
                                                        false, false);
        c = __builtin_amdgcn_wmma_f32_16x16x128_fp8_fp8(a[1], bf[1], (short)0, c,
                                                        false, false);

#pragma unroll
        for (int v = 0; v < 8; ++v) {
            int m = m0 + v + 8 * halfK;
            int n = n0 + lid;
            float accf = c[v];
            if (FIRST)
                accf += noise[(size_t)m * (ROUNDS * HIDDEN) + n];
            int conv = (int)accf;
            if (LAST) {
                Fout[(size_t)m * HIDDEN + n] = (float)(conv % 2);
            } else {
                int p = conv & 1;
                int s = (conv >> 24) & 0x80;
                Xout[(size_t)m * HIDDEN + n] =
                    (unsigned char)(p ? (FP8_ONE | s) : 0);
            }
        }
    }
}


extern "C" void kernel_launch(void* const* d_in, const int* in_sizes, int n_in,
                              void* d_out, int out_size, void* d_ws, size_t ws_size,
                              hipStream_t stream) {
    const int*   nonces   = (const int*)d_in[0];
    const float* matrices = (const float*)d_in[1];
    const float* noise    = (const float*)d_in[2];
    float*       out      = (float*)d_out;

    char* ws = (char*)d_ws;
    unsigned char* mats8 = (unsigned char*)ws;
    unsigned char* xA    = (unsigned char*)(ws + (size_t)ROUNDS * HIDDEN * HIDDEN);
    unsigned char* xB    = xA + (size_t)BATCH * HIDDEN;

    {
        int n = ROUNDS * HIDDEN * HIDDEN;
        mats_to_fp8<<<(n + 255) / 256, 256, 0, stream>>>(matrices, mats8, n);
    }
    unpack_bits_fp8<<<(BATCH * HIDDEN) / 256, 256, 0, stream>>>(nonces, xA);

    unsigned char* cur = xA;
    unsigned char* nxt = xB;
    for (int r = 0; r < ROUNDS; ++r) {
        const unsigned char* M8 = mats8 + (size_t)r * HIDDEN * HIDDEN;
        if (r == 0)
            round_kernel<true, false><<<BATCH / 32, 256, 0, stream>>>(
                cur, M8, noise, nxt, nullptr);
        else if (r == ROUNDS - 1)
            round_kernel<false, true><<<BATCH / 32, 256, 0, stream>>>(
                cur, M8, nullptr, nullptr, out);
        else
            round_kernel<false, false><<<BATCH / 32, 256, 0, stream>>>(
                cur, M8, nullptr, nxt, nullptr);
        unsigned char* tmp = cur; cur = nxt; nxt = tmp;
    }
}
